// kv_cache_multihead_26895085208069
// MI455X (gfx1250) — matrix operand order repaired and checked by reading, not compiled, not run on hardware
//
#include <hip/hip_runtime.h>

typedef __attribute__((ext_vector_type(16))) _Float16 v16h;
typedef __attribute__((ext_vector_type(8)))  _Float16 v8h;
typedef __attribute__((ext_vector_type(8)))  float    v8f;

#define Bv 2
#define Tv 2048
#define Cv 1024
#define Hv 16
#define Dv 64
#define Mrows (Bv*Tv)      /* 4096 */
#define Nqkv  (3*Cv)       /* 3072 */

__device__ inline v8f wmma16(v16h a, v16h b, v8f c) {
  // D = A(16x32 f16) x B(32x16 f16) + C(16x16 f32)
  return __builtin_amdgcn_wmma_f32_16x16x32_f16(
      /*neg_a=*/false, a, /*neg_b=*/false, b,
      /*c_mod=*/(short)0, c, /*reuse_a=*/false, /*reuse_b=*/false);
}

// A fragment (16x32, row-major source, tile origin at `base`):
// lane L: m = L&15, half = L>>4; element e -> K = 16*half + e (contiguous along K)
__device__ inline v16h load_a_frag(const _Float16* base, int ld, int lane) {
  int m = lane & 15, lh = lane >> 4;
  const _Float16* p = base + (size_t)m * ld + lh * 16;
  v8h lo = *(const v8h*)(p);
  v8h hi = *(const v8h*)(p + 8);
  v16h a;
#pragma unroll
  for (int e = 0; e < 8; ++e) { a[e] = lo[e]; a[e + 8] = hi[e]; }
  return a;
}

// B fragment from row-major B^T (an [N x K] array: weight rows / key rows / Vt rows):
// lane L: n = L&15, half = L>>4; element e -> K = 16*half + e (contiguous along K)
__device__ inline v16h load_bt_frag(const _Float16* base, int ld, int lane) {
  int n = lane & 15, lh = lane >> 4;
  const _Float16* p = base + (size_t)n * ld + lh * 16;
  v8h lo = *(const v8h*)(p);
  v8h hi = *(const v8h*)(p + 8);
  v16h b;
#pragma unroll
  for (int e = 0; e < 8; ++e) { b[e] = lo[e]; b[e + 8] = hi[e]; }
  return b;
}

__global__ void f32_to_f16_kernel(const float* __restrict__ in,
                                  _Float16* __restrict__ out, int n) {
  int i = blockIdx.x * blockDim.x + threadIdx.x;
  if (i < n) out[i] = (_Float16)in[i];
}

// ---------------------------------------------------------------------------
// qkv = x @ w_qkv^T, LDS-tiled 128x128 blocks, double-buffered, RoPE epilogue.
// Q,K stored [row, C] f16; V stored transposed [b,h,d,t] f16.
// ---------------------------------------------------------------------------
__global__ __launch_bounds__(256) void qkv_rope_kernel(
    const _Float16* __restrict__ xh, const _Float16* __restrict__ wh,
    _Float16* __restrict__ Qh, _Float16* __restrict__ Kh, _Float16* __restrict__ Vt) {
  __shared__ __align__(16) _Float16 As[2][128 * 40];
  __shared__ __align__(16) _Float16 Bs[2][128 * 40];
  int tid = threadIdx.x, lane = tid & 31, wave = tid >> 5;
  int wm = wave >> 2, wn = wave & 3;       // 2x4 wave grid: 64 rows x 32 cols each
  int row0 = blockIdx.x * 128;
  int n0   = blockIdx.y * 128;

  v8f zero = {};
  v8f acc[4][2];
#pragma unroll
  for (int mi = 0; mi < 4; ++mi)
#pragma unroll
    for (int ni = 0; ni < 2; ++ni) acc[mi][ni] = zero;

  int i = tid >> 1, c = (tid & 1) * 16;    // coop-load: 2 x 8-half chunks / thread / array
  const _Float16* ag = xh + (size_t)(row0 + i) * Cv + c;
  const _Float16* bg = wh + (size_t)(n0 + i) * Cv + c;
  int so = i * 40 + c;                     // LDS slot for this thread

  v8h a0 = *(const v8h*)(ag);
  v8h a1 = *(const v8h*)(ag + 8);
  v8h b0 = *(const v8h*)(bg);
  v8h b1 = *(const v8h*)(bg + 8);
  *(v8h*)&As[0][so] = a0; *(v8h*)&As[0][so + 8] = a1;
  *(v8h*)&Bs[0][so] = b0; *(v8h*)&Bs[0][so + 8] = b1;

  for (int k0 = 0; k0 < Cv; k0 += 32) {
    int cur = (k0 >> 5) & 1;
    bool more = (k0 + 32) < Cv;
    if (more) {                            // stage next tile while this one computes
      a0 = *(const v8h*)(ag + k0 + 32);
      a1 = *(const v8h*)(ag + k0 + 40);
      b0 = *(const v8h*)(bg + k0 + 32);
      b1 = *(const v8h*)(bg + k0 + 40);
    }
    __syncthreads();                       // publishes buf[cur]; readers of old buf done
    v16h af[4];
#pragma unroll
    for (int mi = 0; mi < 4; ++mi)
      af[mi] = load_a_frag(&As[cur][(wm * 64 + mi * 16) * 40], 40, lane);
#pragma unroll
    for (int ni = 0; ni < 2; ++ni) {
      v16h bf = load_bt_frag(&Bs[cur][(wn * 32 + ni * 16) * 40], 40, lane);
#pragma unroll
      for (int mi = 0; mi < 4; ++mi)
        acc[mi][ni] = wmma16(af[mi], bf, acc[mi][ni]);
    }
    if (more) {
      *(v8h*)&As[1 - cur][so] = a0; *(v8h*)&As[1 - cur][so + 8] = a1;
      *(v8h*)&Bs[1 - cur][so] = b0; *(v8h*)&Bs[1 - cur][so + 8] = b1;
    }
  }

  int n_ = lane & 15, lh = lane >> 4;
#pragma unroll
  for (int ni = 0; ni < 2; ++ni) {
    int col = n0 + wn * 32 + ni * 16 + n_;
    int sec = col >> 10;                    // 0=q 1=k 2=v
    int cc  = col & (Cv - 1);
    int h = cc >> 6, d = cc & (Dv - 1);
    float cs = 1.0f, sn = 0.0f;
    if (sec < 2) {                          // rope angles depend on (head, dim) only
      float j = (float)(d >> 1);
      float ang = (float)h * __powf(10000.0f, -j * (1.0f / 32.0f));
      cs = __cosf(ang); sn = __sinf(ang);
    }
#pragma unroll
    for (int mi = 0; mi < 4; ++mi) {
#pragma unroll
      for (int r = 0; r < 8; ++r) {
        float v = acc[mi][ni][r];
        float other = __shfl_xor(v, 1, 32); // rope pair partner (adjacent n lane)
        float o = v;
        if (sec < 2)
          o = ((d & 1) == 0) ? (v * cs - other * sn) : (other * sn + v * cs);
        int row = row0 + wm * 64 + mi * 16 + lh * 8 + r;
        if (sec == 0) {
          Qh[(size_t)row * Cv + cc] = (_Float16)o;
        } else if (sec == 1) {
          Kh[(size_t)row * Cv + cc] = (_Float16)o;
        } else {
          int b = row >> 11, t = row & (Tv - 1);
          Vt[((size_t)(b * Hv + h) * Dv + d) * Tv + t] = (_Float16)o;
        }
      }
    }
  }
}

// ---------------------------------------------------------------------------
// Flash attention. One block = 8 adjacent q-tiles of one (b,h); K/V tiles
// double-buffered in LDS, loaded once per block. Block-uniform s-loop +
// wave-uniform compute guard keeps barriers legal and EXEC all-ones at WMMA.
// ---------------------------------------------------------------------------
__global__ __launch_bounds__(256) void attn_kernel(
    const _Float16* __restrict__ Qh, const _Float16* __restrict__ Kh,
    const _Float16* __restrict__ Vt, _Float16* __restrict__ ctxh) {
  __shared__ __align__(16) _Float16 kt[2][32 * 72];    // K tile: [key 32][d 64] pad 72
  __shared__ __align__(16) _Float16 vtile[2][64 * 40]; // V tile: [d 64][s 32] pad 40
  __shared__ __align__(16) _Float16 ptile[8][16 * 40]; // per-wave P tile (16x32, ld 40)
  int tid = threadIdx.x, lane = tid & 31, wave = tid >> 5;
  int b = blockIdx.x >> 4, h = blockIdx.x & (Hv - 1);
  int qbase = blockIdx.y * 128;
  int q0 = qbase + wave * 16;

  const _Float16* Qb  = Qh + (size_t)b * Tv * Cv + h * Dv;
  const _Float16* Kb  = Kh + (size_t)b * Tv * Cv + h * Dv;
  const _Float16* VtB = Vt + (size_t)(b * Hv + h) * Dv * Tv;

  v16h qa0 = load_a_frag(Qb + (size_t)q0 * Cv + 0,  Cv, lane);
  v16h qa1 = load_a_frag(Qb + (size_t)q0 * Cv + 32, Cv, lane);

  float mrun[8], lrun[8];
  v8f acc0 = {}, acc1 = {}, acc2 = {}, acc3 = {};
#pragma unroll
  for (int r = 0; r < 8; ++r) { mrun[r] = -1e30f; lrun[r] = 0.0f; }
  const float scale = 0.125f;               // 1/sqrt(64)
  int n_ = lane & 15, lh = lane >> 4;
  _Float16* pt = &ptile[wave][0];

  int ki = tid >> 3, kd = (tid & 7) * 8;    // coop K load: 32 rows x 8 chunks
  int vd = tid >> 2, vj = (tid & 3) * 8;    // coop V load: 64 rows x 4 chunks
  int kso = ki * 72 + kd, vso = vd * 40 + vj;
  const int smax = qbase + 128;

  v8h kreg = *(const v8h*)(Kb + (size_t)ki * Cv + kd);
  v8h vreg = *(const v8h*)(VtB + (size_t)vd * Tv + vj);
  *(v8h*)&kt[0][kso] = kreg;
  *(v8h*)&vtile[0][vso] = vreg;

  for (int s0 = 0; s0 < smax; s0 += 32) {
    int cur = (s0 >> 5) & 1;
    bool more = (s0 + 32) < smax;
    if (more) {                             // stage next K/V tile during compute
      kreg = *(const v8h*)(Kb + (size_t)(s0 + 32 + ki) * Cv + kd);
      vreg = *(const v8h*)(VtB + (size_t)vd * Tv + s0 + 32 + vj);
    }
    __syncthreads();
    if (s0 < q0 + 16) {                     // wave-uniform causal skip
      const _Float16* kcur = &kt[cur][0];
      const _Float16* vcur = &vtile[cur][0];
      v8f sc0 = {}, sc1 = {};
      {
        v16h kb;
        kb = load_bt_frag(kcur + 0,            72, lane); sc0 = wmma16(qa0, kb, sc0);
        kb = load_bt_frag(kcur + 32,           72, lane); sc0 = wmma16(qa1, kb, sc0);
        kb = load_bt_frag(kcur + 16 * 72 + 0,  72, lane); sc1 = wmma16(qa0, kb, sc1);
        kb = load_bt_frag(kcur + 16 * 72 + 32, 72, lane); sc1 = wmma16(qa1, kb, sc1);
      }
#pragma unroll
      for (int r = 0; r < 8; ++r) {
        int qrow = q0 + lh * 8 + r;
        float v0 = sc0[r] * scale; if (s0 + n_ > qrow)      v0 = -1e30f;
        float v1 = sc1[r] * scale; if (s0 + 16 + n_ > qrow) v1 = -1e30f;
        float mx = fmaxf(v0, v1);           // row max across the 16 n-lanes
        mx = fmaxf(mx, __shfl_xor(mx, 1, 32));
        mx = fmaxf(mx, __shfl_xor(mx, 2, 32));
        mx = fmaxf(mx, __shfl_xor(mx, 4, 32));
        mx = fmaxf(mx, __shfl_xor(mx, 8, 32));
        float mnew = fmaxf(mrun[r], mx);
        float corr = __expf(mrun[r] - mnew);
        float p0 = __expf(v0 - mnew);
        float p1 = __expf(v1 - mnew);
        float rs = p0 + p1;
        rs += __shfl_xor(rs, 1, 32);
        rs += __shfl_xor(rs, 2, 32);
        rs += __shfl_xor(rs, 4, 32);
        rs += __shfl_xor(rs, 8, 32);
        lrun[r] = lrun[r] * corr + rs;
        mrun[r] = mnew;
        acc0[r] *= corr; acc1[r] *= corr; acc2[r] *= corr; acc3[r] *= corr;
        int prow = lh * 8 + r;              // C-layout -> LDS (row-major 16x32)
        pt[prow * 40 + n_]      = (_Float16)p0;
        pt[prow * 40 + 16 + n_] = (_Float16)p1;
      }
      // DS ops are in-order per wave; wait for P stores, block compiler reordering
      __asm__ volatile("s_wait_dscnt 0" ::: "memory");
      v16h pa = load_a_frag(pt, 40, lane);  // P as A-fragment (16x32)
      v16h vb;
      vb = load_bt_frag(vcur +  0 * 40, 40, lane); acc0 = wmma16(pa, vb, acc0);
      vb = load_bt_frag(vcur + 16 * 40, 40, lane); acc1 = wmma16(pa, vb, acc1);
      vb = load_bt_frag(vcur + 32 * 40, 40, lane); acc2 = wmma16(pa, vb, acc2);
      vb = load_bt_frag(vcur + 48 * 40, 40, lane); acc3 = wmma16(pa, vb, acc3);
    }
    if (more) {
      *(v8h*)&kt[1 - cur][kso] = kreg;
      *(v8h*)&vtile[1 - cur][vso] = vreg;
    }
  }
#pragma unroll
  for (int r = 0; r < 8; ++r) {
    float inv = __builtin_amdgcn_rcpf(lrun[r]);
    size_t row = (size_t)b * Tv + q0 + lh * 8 + r;
    ctxh[row * Cv + h * Dv +  0 + n_] = (_Float16)(acc0[r] * inv);
    ctxh[row * Cv + h * Dv + 16 + n_] = (_Float16)(acc1[r] * inv);
    ctxh[row * Cv + h * Dv + 32 + n_] = (_Float16)(acc2[r] * inv);
    ctxh[row * Cv + h * Dv + 48 + n_] = (_Float16)(acc3[r] * inv);
  }
}

// ---------------------------------------------------------------------------
// out = ctx @ w_out^T (f32 output), LDS-tiled 128x128 blocks, double-buffered.
// ---------------------------------------------------------------------------
__global__ __launch_bounds__(256) void outproj_kernel(
    const _Float16* __restrict__ ctxh, const _Float16* __restrict__ wouth,
    float* __restrict__ out) {
  __shared__ __align__(16) _Float16 As[2][128 * 40];
  __shared__ __align__(16) _Float16 Bs[2][128 * 40];
  int tid = threadIdx.x, lane = tid & 31, wave = tid >> 5;
  int wm = wave >> 2, wn = wave & 3;
  int row0 = blockIdx.x * 128;
  int n0   = blockIdx.y * 128;

  v8f zero = {};
  v8f acc[4][2];
#pragma unroll
  for (int mi = 0; mi < 4; ++mi)
#pragma unroll
    for (int ni = 0; ni < 2; ++ni) acc[mi][ni] = zero;

  int i = tid >> 1, c = (tid & 1) * 16;
  const _Float16* ag = ctxh + (size_t)(row0 + i) * Cv + c;
  const _Float16* bg = wouth + (size_t)(n0 + i) * Cv + c;
  int so = i * 40 + c;

  v8h a0 = *(const v8h*)(ag);
  v8h a1 = *(const v8h*)(ag + 8);
  v8h b0 = *(const v8h*)(bg);
  v8h b1 = *(const v8h*)(bg + 8);
  *(v8h*)&As[0][so] = a0; *(v8h*)&As[0][so + 8] = a1;
  *(v8h*)&Bs[0][so] = b0; *(v8h*)&Bs[0][so + 8] = b1;

  for (int k0 = 0; k0 < Cv; k0 += 32) {
    int cur = (k0 >> 5) & 1;
    bool more = (k0 + 32) < Cv;
    if (more) {
      a0 = *(const v8h*)(ag + k0 + 32);
      a1 = *(const v8h*)(ag + k0 + 40);
      b0 = *(const v8h*)(bg + k0 + 32);
      b1 = *(const v8h*)(bg + k0 + 40);
    }
    __syncthreads();
    v16h af[4];
#pragma unroll
    for (int mi = 0; mi < 4; ++mi)
      af[mi] = load_a_frag(&As[cur][(wm * 64 + mi * 16) * 40], 40, lane);
#pragma unroll
    for (int ni = 0; ni < 2; ++ni) {
      v16h bf = load_bt_frag(&Bs[cur][(wn * 32 + ni * 16) * 40], 40, lane);
#pragma unroll
      for (int mi = 0; mi < 4; ++mi)
        acc[mi][ni] = wmma16(af[mi], bf, acc[mi][ni]);
    }
    if (more) {
      *(v8h*)&As[1 - cur][so] = a0; *(v8h*)&As[1 - cur][so + 8] = a1;
      *(v8h*)&Bs[1 - cur][so] = b0; *(v8h*)&Bs[1 - cur][so + 8] = b1;
    }
  }

  int n_ = lane & 15, lh = lane >> 4;
#pragma unroll
  for (int ni = 0; ni < 2; ++ni)
#pragma unroll
    for (int mi = 0; mi < 4; ++mi)
#pragma unroll
      for (int r = 0; r < 8; ++r) {
        int row = row0 + wm * 64 + mi * 16 + lh * 8 + r;
        out[(size_t)row * Cv + n0 + wn * 32 + ni * 16 + n_] = acc[mi][ni][r];
      }
}

extern "C" void kernel_launch(void* const* d_in, const int* in_sizes, int n_in,
                              void* d_out, int out_size, void* d_ws, size_t ws_size,
                              hipStream_t stream) {
  (void)in_sizes; (void)n_in; (void)out_size; (void)ws_size;
  const float* x    = (const float*)d_in[0];
  const float* wqkv = (const float*)d_in[1];
  const float* wout = (const float*)d_in[2];
  // d_in[3]/d_in[4] (caches) and d_in[5] (start_pos=0) unused: start_pos==0 means
  // k_full/v_full are exactly the freshly-computed k/v.
  float* out = (float*)d_out;

  _Float16* ws = (_Float16*)d_ws;
  const size_t NX  = (size_t)Mrows * Cv;    // 4,194,304
  const size_t NWQ = (size_t)Nqkv * Cv;     // 3,145,728
  const size_t NWO = (size_t)Cv * Cv;       // 1,048,576
  _Float16* xh    = ws;
  _Float16* wqkvh = xh + NX;
  _Float16* wouth = wqkvh + NWQ;
  _Float16* Qh    = wouth + NWO;
  _Float16* Kh    = Qh + NX;
  _Float16* Vt    = Kh + NX;                // transposed [b,h,d,t]
  _Float16* ctxh  = Vt + NX;

  f32_to_f16_kernel<<<(int)((NX  + 255) / 256), 256, 0, stream>>>(x,    xh,    (int)NX);
  f32_to_f16_kernel<<<(int)((NWQ + 255) / 256), 256, 0, stream>>>(wqkv, wqkvh, (int)NWQ);
  f32_to_f16_kernel<<<(int)((NWO + 255) / 256), 256, 0, stream>>>(wout, wouth, (int)NWO);

  qkv_rope_kernel<<<dim3(Mrows / 128, Nqkv / 128), 256, 0, stream>>>(xh, wqkvh, Qh, Kh, Vt);
  attn_kernel<<<dim3(Bv * Hv, Tv / 128), 256, 0, stream>>>(Qh, Kh, Vt, ctxh);
  outproj_kernel<<<dim3(Mrows / 128, Cv / 128), 256, 0, stream>>>(ctxh, wouth, out);
}
